// GRINMixtureOfExpertFFN_5909874999574
// MI455X (gfx1250) — matrix operand order repaired and checked by reading, not compiled, not run on hardware
//
#include <hip/hip_runtime.h>

#define T_TOK 2048
#define HID   1024
#define FFN_  2048
#define NEXP  8
#define LDS_PITCH (HID + 8)   // bf16 elements; 16B-aligned rows, conflict-free reads

typedef __attribute__((ext_vector_type(16))) __bf16 v16bf;
typedef __attribute__((ext_vector_type(8)))  float  v8f;

union Frag16 { v16bf v; uint4 q[2]; };   // 32B: one WMMA A/B bf16 fragment per lane

// ---------------- init: zero output + expert counters ----------------
__global__ void k_init(float* __restrict__ out, int* __restrict__ counts) {
  int i = blockIdx.x * blockDim.x + threadIdx.x;
  if (i < T_TOK * HID) out[i] = 0.0f;
  if (i < NEXP) counts[i] = 0;
}

// ---------------- vectorized fp32 -> bf16 (4 elems/thread) ----------------
__global__ void k_cvt_v4(const float* __restrict__ src, __bf16* __restrict__ dst, int n4) {
  int i = blockIdx.x * blockDim.x + threadIdx.x;
  if (i < n4) {
    float4 f = ((const float4*)src)[i];
    union { __bf16 h[4]; uint2 u; } o;
    o.h[0] = (__bf16)f.x; o.h[1] = (__bf16)f.y; o.h[2] = (__bf16)f.z; o.h[3] = (__bf16)f.w;
    ((uint2*)dst)[i] = o.u;
  }
}

// ---------------- router: softmax -> top2 -> renorm -> compact lists ----------------
__global__ __launch_bounds__(256) void k_router(
    const float* __restrict__ x, const float* __restrict__ gw,
    int* __restrict__ counts, int* __restrict__ tok, float* __restrict__ tokw) {
  int t = blockIdx.x * 256 + threadIdx.x;
  if (t >= T_TOK) return;
  float acc[NEXP];
#pragma unroll
  for (int e = 0; e < NEXP; ++e) acc[e] = 0.0f;
  const float* xp = x + (size_t)t * HID;
  for (int k = 0; k < HID; ++k) {
    float xv = xp[k];
#pragma unroll
    for (int e = 0; e < NEXP; ++e) acc[e] += xv * gw[e * HID + k];
  }
  float m = acc[0];
#pragma unroll
  for (int e = 1; e < NEXP; ++e) m = fmaxf(m, acc[e]);
  float p[NEXP]; float s = 0.0f;
#pragma unroll
  for (int e = 0; e < NEXP; ++e) { p[e] = __expf(acc[e] - m); s += p[e]; }
  float inv = 1.0f / s;
#pragma unroll
  for (int e = 0; e < NEXP; ++e) p[e] *= inv;
  int i1 = 0;
#pragma unroll
  for (int e = 1; e < NEXP; ++e) if (p[e] > p[i1]) i1 = e;
  int i2 = (i1 == 0) ? 1 : 0;
#pragma unroll
  for (int e = 0; e < NEXP; ++e) if (e != i1 && p[e] > p[i2]) i2 = e;
  float wsum = p[i1] + p[i2];
  float wa = p[i1] / wsum, wb = p[i2] / wsum;
  int p1 = atomicAdd(&counts[i1], 1);
  tok[i1 * T_TOK + p1] = t;  tokw[i1 * T_TOK + p1] = wa;
  int p2 = atomicAdd(&counts[i2], 1);
  tok[i2 * T_TOK + p2] = t;  tokw[i2 * T_TOK + p2] = wb;
}

// ---------------- expert stage 1: h = silu(X w1^T) * (X w3^T), bf16 WMMA ----------------
// grid: (FFN_/128, T_TOK/16); block 256 (8 waves); wave n-tile = 16 cols of FFN.
// Weights already bf16 (pre-converted): k-loop body = pure loads + WMMA.
__global__ __launch_bounds__(256) void k_expert_h(
    const __bf16* __restrict__ w1b, const __bf16* __restrict__ w3b,
    const __bf16* __restrict__ xbf, const int* __restrict__ counts,
    const int* __restrict__ tok, __bf16* __restrict__ h, int e) {
  __shared__ __bf16 xs[16 * LDS_PITCH];
  const int tid   = threadIdx.x;
  const int cnt   = counts[e];
  const int mbase = blockIdx.y * 16;
  if (mbase >= cnt) return;

  // Stage 16 gathered token rows (bf16) into LDS, 16B vectors, padded pitch.
  {
    int row = tid >> 4;            // 0..15
    int seg = tid & 15;            // 0..15, 64 bf16 each
    int idx = mbase + row; if (idx > cnt - 1) idx = cnt - 1;
    int t = tok[e * T_TOK + idx];
    const uint4* src = (const uint4*)(xbf + (size_t)t * HID) + seg * 8;
    uint4* dst = (uint4*)(xs + row * LDS_PITCH) + seg * 8;
#pragma unroll
    for (int i = 0; i < 8; ++i) dst[i] = src[i];
  }
  __syncthreads();

  const int wave = tid >> 5;
  const int lane = tid & 31;
  const int half = lane >> 4;        // K-half select per WMMA layout
  const int mrow = lane & 15;        // A row / B,C col within tile
  const int ncol = blockIdx.x * 128 + wave * 16 + mrow;   // FFN column

  const __bf16* w1p = w1b + (size_t)ncol * HID;
  const __bf16* w3p = w3b + (size_t)ncol * HID;

  v8f acc1 = {}; v8f acc3 = {};
  for (int kb = 0; kb < HID; kb += 32) {
    Frag16 a, b1, b3;
    const __bf16* ar = xs + mrow * LDS_PITCH + kb + half * 16;
    a.q[0] = *(const uint4*)(ar);
    a.q[1] = *(const uint4*)(ar + 8);
    const __bf16* b1p = w1p + kb + half * 16;      // 16 consecutive bf16 = 32B
    b1.q[0] = *(const uint4*)(b1p);
    b1.q[1] = *(const uint4*)(b1p + 8);
    const __bf16* b3p = w3p + kb + half * 16;
    b3.q[0] = *(const uint4*)(b3p);
    b3.q[1] = *(const uint4*)(b3p + 8);
    acc1 = __builtin_amdgcn_wmma_f32_16x16x32_bf16(false, a.v, false, b1.v, (short)0, acc1, false, false);
    acc3 = __builtin_amdgcn_wmma_f32_16x16x32_bf16(false, a.v, false, b3.v, (short)0, acc3, false, false);
  }

  // SwiGLU epilogue (fast rcp, no IEEE div) + bf16 store (C layout: M = i + 8*half, N = mrow).
#pragma unroll
  for (int i = 0; i < 8; ++i) {
    float g   = acc1[i];
    float sig = __builtin_amdgcn_rcpf(1.0f + __expf(-g));
    float hv  = g * sig * acc3[i];
    int m = half * 8 + i;
    h[(size_t)(mbase + m) * FFN_ + ncol] = (__bf16)hv;
  }
}

// ---------------- expert stage 2: out[tok] += w * (h w2^T) ----------------
// grid: (HID/128, T_TOK/16); block 256 (8 waves); wave n-tile = 16 cols of HID.
__global__ __launch_bounds__(256) void k_expert_out(
    const __bf16* __restrict__ w2b, const __bf16* __restrict__ h,
    const int* __restrict__ counts, const int* __restrict__ tok,
    const float* __restrict__ tokw, float* __restrict__ out, int e) {
  const int tid   = threadIdx.x;
  const int cnt   = counts[e];
  const int mbase = blockIdx.y * 16;
  if (mbase >= cnt) return;

  const int wave = tid >> 5;
  const int lane = tid & 31;
  const int half = lane >> 4;
  const int mrow = lane & 15;
  const int ncol = blockIdx.x * 128 + wave * 16 + mrow;   // HID column

  const __bf16* w2p = w2b + (size_t)ncol * FFN_;
  const __bf16* hp  = h + (size_t)(mbase + mrow) * FFN_;

  v8f acc = {};
  for (int kb = 0; kb < FFN_; kb += 32) {
    Frag16 a, b;
    const __bf16* ar = hp + kb + half * 16;
    a.q[0] = *(const uint4*)(ar);
    a.q[1] = *(const uint4*)(ar + 8);
    const __bf16* bp = w2p + kb + half * 16;
    b.q[0] = *(const uint4*)(bp);
    b.q[1] = *(const uint4*)(bp + 8);
    acc = __builtin_amdgcn_wmma_f32_16x16x32_bf16(false, a.v, false, b.v, (short)0, acc, false, false);
  }

  // Scatter-accumulate weighted rows back to tokens (unique tokens per expert => no races).
#pragma unroll
  for (int i = 0; i < 8; ++i) {
    int m = half * 8 + i;
    int gidx = mbase + m;
    if (gidx < cnt) {
      int   t  = tok[e * T_TOK + gidx];
      float wt = tokw[e * T_TOK + gidx];
      size_t o = (size_t)t * HID + ncol;
      out[o] += wt * acc[i];
    }
  }
}

extern "C" void kernel_launch(void* const* d_in, const int* in_sizes, int n_in,
                              void* d_out, int out_size, void* d_ws, size_t ws_size,
                              hipStream_t stream) {
  const float* x  = (const float*)d_in[0];
  const float* gw = (const float*)d_in[1];
  const float* w1 = (const float*)d_in[2];
  const float* w3 = (const float*)d_in[3];
  const float* w2 = (const float*)d_in[4];
  float* out = (float*)d_out;

  char* ws = (char*)d_ws;
  int*    counts = (int*)(ws);                                   // 8 ints
  int*    tok    = (int*)(ws + 1024);                            // E*T ints   (64 KB)
  float*  tokw   = (float*)(ws + 1024 + 65536);                  // E*T floats (64 KB)
  __bf16* xbf    = (__bf16*)(ws + 132096);                       // T*H bf16   (4 MB)
  __bf16* hbuf   = (__bf16*)(ws + 132096 + 4194304);             // T*F bf16   (8 MB)
  __bf16* w1bf   = (__bf16*)(ws + 132096 + 4194304 + 8388608);   // F*H bf16   (4 MB, per-expert)
  __bf16* w3bf   = w1bf + (size_t)FFN_ * HID;                    // F*H bf16   (4 MB, per-expert)
  __bf16* w2bf   = w3bf + (size_t)FFN_ * HID;                    // H*F bf16   (4 MB, per-expert)

  const int N4W = (FFN_ * HID) / 4;   // 524288 float4s per expert weight matrix

  k_init  <<<8192, 256, 0, stream>>>(out, counts);
  k_cvt_v4<<<(T_TOK * HID / 4 + 255) / 256, 256, 0, stream>>>(x, xbf, T_TOK * HID / 4);
  k_router<<<8, 256, 0, stream>>>(x, gw, counts, tok, tokw);

  for (int e = 0; e < NEXP; ++e) {
    const float* w1e = w1 + (size_t)e * FFN_ * HID;
    const float* w3e = w3 + (size_t)e * FFN_ * HID;
    const float* w2e = w2 + (size_t)e * HID * FFN_;
    // Pre-convert this expert's weights to bf16 (reused by ~32 M-tiles, L2-resident).
    k_cvt_v4<<<(N4W + 255) / 256, 256, 0, stream>>>(w1e, w1bf, N4W);
    k_cvt_v4<<<(N4W + 255) / 256, 256, 0, stream>>>(w3e, w3bf, N4W);
    k_cvt_v4<<<(N4W + 255) / 256, 256, 0, stream>>>(w2e, w2bf, N4W);

    k_expert_h  <<<dim3(FFN_ / 128, T_TOK / 16), 256, 0, stream>>>(w1bf, w3bf, xbf, counts, tok, hbuf, e);
    k_expert_out<<<dim3(HID  / 128, T_TOK / 16), 256, 0, stream>>>(w2bf, hbuf, counts, tok, tokw, out, e);
  }
}
